// FilmBlock_18425409700220
// MI455X (gfx1250) — matrix operand order repaired and checked by reading, not compiled, not run on hardware
//
#include <hip/hip_runtime.h>

typedef __bf16 bf16;
typedef __attribute__((ext_vector_type(16))) __bf16 v16bf;
typedef __attribute__((ext_vector_type(8)))  __bf16 bf16x8;
typedef __attribute__((ext_vector_type(8)))  float  v8f;
typedef __attribute__((ext_vector_type(4)))  unsigned int u32x4;
typedef __attribute__((ext_vector_type(8)))  int i32x8;
typedef __attribute__((ext_vector_type(4)))  int i32x4;

#define B_    16
#define H_    64
#define W_    64
#define C_    256
#define CINX  128
#define CINY  512
#define HP    66              // padded spatial extent

__device__ __forceinline__ float gelu_exact(float x) {
    return 0.5f * x * (1.0f + erff(x * 0.70710678118654752440f));
}

__device__ __forceinline__ v16bf combine16(bf16x8 lo, bf16x8 hi) {
    union { v16bf v; bf16x8 h[2]; } u;
    u.h[0] = lo; u.h[1] = hi;
    return u.v;
}

// ---------------- prep kernels ----------------

// x [B,CinX,H,W] f32 -> x padded NHWC bf16 [B,66,66,CINX] (zero halo)
__global__ __launch_bounds__(256) void prep_x(const float* __restrict__ x,
                                              bf16* __restrict__ xp) {
    int i = blockIdx.x * 256 + threadIdx.x;          // 0 .. B*66*66*CINX-1
    int ci   = i & (CINX - 1);
    int rest = i >> 7;
    int wp = rest % HP;
    int t2 = rest / HP;
    int hp = t2 % HP;
    int b  = t2 / HP;
    float v = 0.0f;
    if (hp >= 1 && hp <= H_ && wp >= 1 && wp <= W_)
        v = x[(((size_t)b * CINX + ci) * H_ + (hp - 1)) * W_ + (wp - 1)];
    xp[i] = (bf16)v;
}

// conv_w [O,CinX,3,3] f32 -> wt [9][O][CinX] bf16
__global__ __launch_bounds__(256) void prep_w(const float* __restrict__ cw,
                                              bf16* __restrict__ wt) {
    int i = blockIdx.x * 256 + threadIdx.x;          // 0 .. 9*C_*CINX-1
    int ci  = i & (CINX - 1);
    int o   = (i >> 7) & (C_ - 1);
    int tap = i >> 15;
    int kh = tap / 3, kw = tap % 3;
    wt[i] = (bf16)cw[(((size_t)o * CINX + ci) * 3 + kh) * 3 + kw];
}

// Wix, Wo [C,C] f32 -> bf16 (layout unchanged)
__global__ __launch_bounds__(256) void prep_lin(const float* __restrict__ wix,
                                                const float* __restrict__ wo,
                                                bf16* __restrict__ wixb,
                                                bf16* __restrict__ wob) {
    int i = blockIdx.x * 256 + threadIdx.x;          // 0 .. 2*C_*C_-1
    if (i < C_ * C_) wixb[i] = (bf16)wix[i];
    else             wob[i - C_ * C_] = (bf16)wo[i - C_ * C_];
}

// ---------------- y branch (tiny) ----------------
__global__ __launch_bounds__(256) void y_branch(
    const float* __restrict__ y,   const float* __restrict__ w1, const float* __restrict__ b1,
    const float* __restrict__ lyg, const float* __restrict__ lyb,
    const float* __restrict__ Wiy, const float* __restrict__ biy,
    const float* __restrict__ Ww,  const float* __restrict__ bw,
    const float* __restrict__ Wb,  const float* __restrict__ bb,
    float* __restrict__ yw, float* __restrict__ yb) {
    __shared__ float s1[256], s2[256], sv[256];
    const int b = blockIdx.x, c = threadIdx.x;

    float yl = b1[c];
    for (int k = 0; k < CINY; ++k) yl = fmaf(y[b * CINY + k], w1[c * CINY + k], yl);

    s1[c] = yl; s2[c] = yl * yl; __syncthreads();
    for (int off = 128; off > 0; off >>= 1) {
        if (c < off) { s1[c] += s1[c + off]; s2[c] += s2[c + off]; }
        __syncthreads();
    }
    float mu  = s1[0] * (1.0f / 256.0f);
    float var = s2[0] * (1.0f / 256.0f) - mu * mu;
    float yn  = (yl - mu) * rsqrtf(var + 1e-5f) * lyg[c] + lyb[c];
    __syncthreads();
    sv[c] = yn; __syncthreads();

    float tv = biy[c];
    for (int k = 0; k < C_; ++k) tv = fmaf(Wiy[c * C_ + k], sv[k], tv);
    tv = gelu_exact(tv);
    __syncthreads();
    sv[c] = tv; __syncthreads();

    float a = bw[c], d = bb[c];
    for (int k = 0; k < C_; ++k) {
        float t = sv[k];
        a = fmaf(Ww[c * C_ + k], t, a);
        d = fmaf(Wb[c * C_ + k], t, d);
    }
    yw[b * C_ + c] = a;
    yb[b * C_ + c] = d;
}

// ---------------- conv stem + fused channel LayerNorm ----------------
// One workgroup per (b,h): M = 64 pixels (w), N = 256 channels, K = 9*128.
// Halo-padded input => branch-free inner loop (pure load -> WMMA).
__global__ __launch_bounds__(256) void conv_stem_ln(
    const bf16* __restrict__ xp,     // [B,66,66,CINX]
    const bf16* __restrict__ wt,     // [9][C][CINX]
    const float* __restrict__ cb,
    const float* __restrict__ lng, const float* __restrict__ lnb,
    float* __restrict__ out_nchw,    // shortcut -> d_out
    bf16* __restrict__ xnb) {        // post-LN [B*H*W, C] bf16
    __shared__ float ps[64][4], pq[64][4];
    const int bh = blockIdx.x;
    const int b = bh >> 6, h = bh & 63;
    const int wave = threadIdx.x >> 5, lane = threadIdx.x & 31;
    const int mi = wave & 3, ni = wave >> 2;
    const int hg = lane >> 4, l15 = lane & 15;

    v8f acc[4];
    #pragma unroll
    for (int s = 0; s < 4; ++s)
        #pragma unroll
        for (int i = 0; i < 8; ++i) acc[s][i] = 0.0f;

    const int wpix = mi * 16 + l15;   // A-fragment row -> w coordinate
    #pragma unroll
    for (int kh = 0; kh < 3; ++kh) {
        #pragma unroll
        for (int kw = 0; kw < 3; ++kw) {
            const bf16* arow = xp + ((size_t)(b * HP + h + kh) * HP + wpix + kw) * CINX;
            const int tap = kh * 3 + kw;
            #pragma unroll
            for (int kc = 0; kc < 4; ++kc) {      // 4 x K32 chunks over CinX=128
                bf16x8 lo = *(const bf16x8*)(arow + kc * 32 + hg * 16);
                bf16x8 hi = *(const bf16x8*)(arow + kc * 32 + hg * 16 + 8);
                const v16bf a = combine16(lo, hi);
                #pragma unroll
                for (int s = 0; s < 4; ++s) {
                    const int o = ni * 64 + s * 16 + l15;
                    const v16bf bm = *(const v16bf*)(wt + ((size_t)(tap * C_ + o)) * CINX +
                                                     kc * 32 + hg * 16);
                    acc[s] = __builtin_amdgcn_wmma_f32_16x16x32_bf16(
                        false, a, false, bm, (short)0, acc[s], false, false);
                }
            }
        }
    }

    // ---- fused LayerNorm over channels ----
    float cbv[4];
    #pragma unroll
    for (int s = 0; s < 4; ++s) cbv[s] = cb[ni * 64 + s * 16 + l15];

    // per-row partials over this wave's 64-channel block
    #pragma unroll
    for (int r = 0; r < 8; ++r) {
        float sm = 0.0f, sq = 0.0f;
        #pragma unroll
        for (int s = 0; s < 4; ++s) {
            const float v = acc[s][r] + cbv[s];
            sm += v; sq += v * v;
        }
        #pragma unroll
        for (int m = 1; m < 16; m <<= 1) {       // reduce across 16-lane column group
            sm += __shfl_xor(sm, m, 32);
            sq += __shfl_xor(sq, m, 32);
        }
        if (l15 == 0) {
            const int row = mi * 16 + hg * 8 + r;
            ps[row][ni] = sm;
            pq[row][ni] = sq;
        }
    }
    __syncthreads();

    float mu[8], rsv[8];
    #pragma unroll
    for (int r = 0; r < 8; ++r) {
        const int row = mi * 16 + hg * 8 + r;
        const float sm = ps[row][0] + ps[row][1] + ps[row][2] + ps[row][3];
        const float sq = pq[row][0] + pq[row][1] + pq[row][2] + pq[row][3];
        const float m0 = sm * (1.0f / 256.0f);
        mu[r]  = m0;
        rsv[r] = rsqrtf(sq * (1.0f / 256.0f) - m0 * m0 + 1e-5f);
    }

    // epilogue: write NCHW shortcut (fp32) + post-LN NHWC bf16
    #pragma unroll
    for (int s = 0; s < 4; ++s) {
        const int o = ni * 64 + s * 16 + l15;
        const float lg = lng[o], lb = lnb[o];
        #pragma unroll
        for (int r = 0; r < 8; ++r) {
            const int m = mi * 16 + hg * 8 + r;    // w coordinate
            const float v = acc[s][r] + cbv[s];
            out_nchw[(((size_t)b * C_ + o) * H_ + h) * W_ + m] = v;
            xnb[((size_t)(b * H_ + h) * W_ + m) * C_ + o] =
                (bf16)((v - mu[r]) * rsv[r] * lg + lb);
        }
    }
}

// ---------------- fused GEMM -> GELU -> FiLM -> GEMM -> +shortcut ----------------
__global__ __launch_bounds__(256) void film_mlp(
    const bf16* __restrict__ xn,    // [B*H*W, C] post-LN bf16
    const bf16* __restrict__ wixb,  // [C,C]
    const bf16* __restrict__ wob,   // [C,C]
    const float* __restrict__ bix,  const float* __restrict__ bo,
    const float* __restrict__ yw,   const float* __restrict__ yb,
    float* __restrict__ out_nchw) {
    __shared__ bf16 tile[64 * 264];               // stride 264: TDM pad gives it for free
    const int bh = blockIdx.x;
    const int b = bh >> 6, h = bh & 63;
    const size_t pixbase = (size_t)(b * H_ + h) * W_;
    const int t = threadIdx.x;

    // --- TDM: DMA the 64x256 bf16 tile into LDS with +16B row padding ---
    if (t < 32) {                                  // wave 0 issues the descriptor
        const unsigned lds_base = (unsigned)(uintptr_t)tile;   // low 32 bits = LDS offset
        const unsigned long long ga =
            (unsigned long long)(uintptr_t)(xn + pixbase * C_);
        u32x4 g0;
        g0[0] = 1u;                                             // count=1 (valid D#)
        g0[1] = lds_base;                                       // lds_addr
        g0[2] = (unsigned)(ga & 0xFFFFFFFFu);                   // global_addr[31:0]
        g0[3] = (unsigned)((ga >> 32) & 0x01FFFFFFu) | (2u << 30); // addr[56:32] | type=2
        i32x8 g1;
        g1[0] = (1 << 16)            // data_size = 2 bytes
              | (1 << 20)            // pad_enable
              | (6 << 22)            // pad_interval: 128 DWORDs (= one 512B row)
              | (3 << 25);           // pad_amount: 4 DWORDs (16B) -> row stride 264 bf16
        g1[1] = (int)(256u << 16);   // tensor_dim0[15:0] = 256
        g1[2] = 0;                   // tensor_dim0[31:16]=0, tensor_dim1[15:0]=0
        g1[3] = (int)(1u | (256u << 16)); // tensor_dim1[31:16]=1 (65536 rows), tile_dim0=256
        g1[4] = 64;                  // tile_dim1=64 rows, tile_dim2=0
        g1[5] = 256;                 // tensor_dim0_stride[31:0] = 256 elements
        g1[6] = 0;
        g1[7] = 0;
        i32x4 z4 = {0, 0, 0, 0};
        i32x8 z8 = {0, 0, 0, 0, 0, 0, 0, 0};
        __builtin_amdgcn_tensor_load_to_lds(g0, g1, z4, z4, z8, 0);
        __builtin_amdgcn_s_wait_tensorcnt(0);
    }
    __syncthreads();

    const int wave = t >> 5, lane = t & 31;
    const int mi = wave & 3, ni = wave >> 2;
    const int hg = lane >> 4, l15 = lane & 15;
    const int arow = mi * 16 + l15;

    v8f acc[4];
    #pragma unroll
    for (int s = 0; s < 4; ++s)
        #pragma unroll
        for (int i = 0; i < 8; ++i) acc[s][i] = 0.0f;

    // GEMM1: t = xn @ Wix^T
    #pragma unroll
    for (int ks = 0; ks < 8; ++ks) {
        bf16x8 lo = *(const bf16x8*)(tile + arow * 264 + ks * 32 + hg * 16);
        bf16x8 hi = *(const bf16x8*)(tile + arow * 264 + ks * 32 + hg * 16 + 8);
        const v16bf a = combine16(lo, hi);
        #pragma unroll
        for (int s = 0; s < 4; ++s) {
            const int o = ni * 64 + s * 16 + l15;
            const v16bf bm = *(const v16bf*)(wixb + (size_t)o * C_ + ks * 32 + hg * 16);
            acc[s] = __builtin_amdgcn_wmma_f32_16x16x32_bf16(
                false, a, false, bm, (short)0, acc[s], false, false);
        }
    }

    // epilogue1: GELU + FiLM, hold as bf16
    bf16 gv[4][8];
    #pragma unroll
    for (int s = 0; s < 4; ++s) {
        const int o = ni * 64 + s * 16 + l15;
        const float bi = bix[o];
        const float fw = yw[b * C_ + o];
        const float fb = yb[b * C_ + o];
        #pragma unroll
        for (int r = 0; r < 8; ++r) {
            const float v = gelu_exact(acc[s][r] + bi);
            gv[s][r] = (bf16)(v * fw + fb);
        }
    }
    __syncthreads();                               // all GEMM1 tile reads done
    #pragma unroll
    for (int s = 0; s < 4; ++s) {
        const int o = ni * 64 + s * 16 + l15;
        #pragma unroll
        for (int r = 0; r < 8; ++r) {
            const int m = mi * 16 + hg * 8 + r;
            tile[m * 264 + o] = gv[s][r];
        }
    }
    __syncthreads();

    // GEMM2: out = g @ Wo^T
    v8f acc2[4];
    #pragma unroll
    for (int s = 0; s < 4; ++s)
        #pragma unroll
        for (int i = 0; i < 8; ++i) acc2[s][i] = 0.0f;
    #pragma unroll
    for (int ks = 0; ks < 8; ++ks) {
        bf16x8 lo = *(const bf16x8*)(tile + arow * 264 + ks * 32 + hg * 16);
        bf16x8 hi = *(const bf16x8*)(tile + arow * 264 + ks * 32 + hg * 16 + 8);
        const v16bf a = combine16(lo, hi);
        #pragma unroll
        for (int s = 0; s < 4; ++s) {
            const int o = ni * 64 + s * 16 + l15;
            const v16bf bm = *(const v16bf*)(wob + (size_t)o * C_ + ks * 32 + hg * 16);
            acc2[s] = __builtin_amdgcn_wmma_f32_16x16x32_bf16(
                false, a, false, bm, (short)0, acc2[s], false, false);
        }
    }

    // epilogue2: + bo + shortcut (RMW d_out, contiguous 8 floats per lane)
    #pragma unroll
    for (int s = 0; s < 4; ++s) {
        const int o = ni * 64 + s * 16 + l15;
        const float bov = bo[o];
        #pragma unroll
        for (int r = 0; r < 8; ++r) {
            const int m = mi * 16 + hg * 8 + r;
            const size_t oi = (((size_t)b * C_ + o) * H_ + h) * W_ + m;
            out_nchw[oi] = out_nchw[oi] + acc2[s][r] + bov;
        }
    }
}

// ---------------- launch ----------------
extern "C" void kernel_launch(void* const* d_in, const int* in_sizes, int n_in,
                              void* d_out, int out_size, void* d_ws, size_t ws_size,
                              hipStream_t stream) {
    (void)in_sizes; (void)n_in; (void)out_size; (void)ws_size;
    const float* x      = (const float*)d_in[0];
    const float* y      = (const float*)d_in[1];
    const float* conv_w = (const float*)d_in[2];
    const float* conv_b = (const float*)d_in[3];
    const float* w1     = (const float*)d_in[4];
    const float* b1     = (const float*)d_in[5];
    const float* ln_x_g = (const float*)d_in[6];
    const float* ln_x_b = (const float*)d_in[7];
    const float* ln_y_g = (const float*)d_in[8];
    const float* ln_y_b = (const float*)d_in[9];
    const float* Wix    = (const float*)d_in[10];
    const float* bix    = (const float*)d_in[11];
    const float* Wiy    = (const float*)d_in[12];
    const float* biy    = (const float*)d_in[13];
    const float* Ww     = (const float*)d_in[14];
    const float* bw     = (const float*)d_in[15];
    const float* Wb     = (const float*)d_in[16];
    const float* bb     = (const float*)d_in[17];
    const float* Wo     = (const float*)d_in[18];
    const float* bo     = (const float*)d_in[19];
    float* out = (float*)d_out;

    // workspace layout (bytes, 256-aligned)
    char* ws = (char*)d_ws;
    bf16*  xp   = (bf16*)(ws);                      // padded x: 16*66*66*128*2 = 17,842,176
    bf16*  wt   = (bf16*)(ws + 17842176);           // 589,824
    bf16*  wixb = (bf16*)(ws + 18432000);           // 131,072
    bf16*  wob  = (bf16*)(ws + 18563072);           // 131,072
    float* ywp  = (float*)(ws + 18694144);          // 16,384
    float* ybp  = (float*)(ws + 18710528);          // 16,384
    bf16*  xnb  = (bf16*)(ws + 18726912);           // 33,554,432  (total ~52.3 MB)

    prep_x  <<<34848, 256, 0, stream>>>(x, xp);     // 16*66*66*128 / 256
    prep_w  <<<1152,  256, 0, stream>>>(conv_w, wt);
    prep_lin<<<512,   256, 0, stream>>>(Wix, Wo, wixb, wob);
    y_branch<<<16,    256, 0, stream>>>(y, w1, b1, ln_y_g, ln_y_b,
                                        Wiy, biy, Ww, bw, Wb, bb, ywp, ybp);
    conv_stem_ln<<<B_ * H_, 256, 0, stream>>>(xp, wt, conv_b, ln_x_g, ln_x_b, out, xnb);
    film_mlp<<<B_ * H_, 256, 0, stream>>>(xnb, wixb, wob, bix, bo, ywp, ybp, out);
}
